// SolarLSTM_20804821582527
// MI455X (gfx1250) — matrix operand order repaired and checked by reading, compiled, not run on hardware
//
#include <hip/hip_runtime.h>

#define BQ  8192
#define TT  512
#define INQ 8
#define HQ  64
#define CH  32
#define NC  (TT / CH)

typedef __attribute__((ext_vector_type(16))) _Float16 v16h;
typedef __attribute__((ext_vector_type(8)))  float    v8f;
typedef __attribute__((ext_vector_type(4)))  float    f32x4;
typedef __attribute__((ext_vector_type(2)))  float    f32x2;

union AF { v16h v; f32x4 q[2]; _Float16 e[16]; };

// gfx1250 has a native V_TANH_F32 transcendental op; use it if the builtin exists.
#if __has_builtin(__builtin_amdgcn_tanhf)
__device__ __forceinline__ float tanh_hw(float x) { return __builtin_amdgcn_tanhf(x); }
#elif __has_builtin(__builtin_amdgcn_tanh_f32)
__device__ __forceinline__ float tanh_hw(float x) { return __builtin_amdgcn_tanh_f32(x); }
#else
__device__ __forceinline__ float tanh_hw(float x) {
  float e = __expf(2.f * x);
  return (e - 1.f) / (e + 1.f);
}
#endif
// sigmoid(x) = 0.5 * (1 + tanh(x/2)) : one trans op, no divide
__device__ __forceinline__ float sigm(float x) { return fmaf(0.5f, tanh_hw(0.5f * x), 0.5f); }

// LDS layout (bytes):
//   [0     , 16384)  xbuf0  : 16 rows x 32 t x 8 f32
//   [16384 , 32768)  xbuf1
//   [32768 , 49152)  gates  : 4 gates x 64 n x 16 m f32
//   [49152 , 51200)  h      : 16 m x 64 k f16
//   [51200 , 51264)  outred : 16 f32
//   [0     , 49152)  bstage : 96 k x 256 n f16 (init only; overlaps xbuf+gates)
__global__ __launch_bounds__(128)
void solar_lstm_wmma(const float* __restrict__ x,
                     const float* __restrict__ W_ih,
                     const float* __restrict__ W_hh,
                     const float* __restrict__ b_ih,
                     const float* __restrict__ b_hh,
                     const float* __restrict__ W_fc,
                     const float* __restrict__ b_fc,
                     float* __restrict__ out)
{
  __shared__ __align__(16) char smem[51264];

  _Float16* bstage   = (_Float16*)smem;
  float*    gate_lds = (float*)(smem + 32768);
  _Float16* h_lds    = (_Float16*)(smem + 49152);
  float*    outred   = (float*)(smem + 51200);

  const int tid  = threadIdx.x;
  const int wid  = tid >> 5;     // wave id 0..3 == gate type (i,f,g,o)
  const int lane = tid & 31;
  const int b0   = blockIdx.x * 16;

  const unsigned ldsbase = (unsigned)(size_t)(void*)smem;

  // ---------------- init: stage combined weights B[k][n] f16 (k<96, n<256) --
  // k in [0,64): W_hh[n][k]; k in [64,72): W_ih[n][k-64]; k==72: b_ih[n]+b_hh[n]
  for (int i = tid; i < 96 * 256; i += 128) {
    int k = i >> 8, n = i & 255;
    float v = 0.f;
    if (k < 64)       v = W_hh[n * 64 + k];
    else if (k < 72)  v = W_ih[n * 8 + (k - 64)];
    else if (k == 72) v = b_ih[n] + b_hh[n];
    bstage[k * 256 + n] = (_Float16)v;
  }
  __syncthreads();

  // B fragments into registers: 3 K-chunks x 4 N-tiles.
  // 16-bit B 32x16 layout: lane holds N = lane&15; elem j -> K = (lane<16?0:8) + j + (j<8?0:8)
  v16h Bf[3][4];
  {
    const int nb    = lane & 15;
    const int khalf = (lane < 16) ? 0 : 8;
    #pragma unroll
    for (int kc = 0; kc < 3; ++kc)
      #pragma unroll
      for (int nt = 0; nt < 4; ++nt) {
        AF t;
        const int nabs = wid * 64 + nt * 16 + nb;
        #pragma unroll
        for (int j = 0; j < 16; ++j)
          t.e[j] = bstage[(kc * 32 + khalf + j + (j < 8 ? 0 : 8)) * 256 + nabs];
        Bf[kc][nt] = t.v;
      }
  }
  // zero initial hidden state (region does not overlap bstage)
  for (int i = tid; i < 16 * 64; i += 128) h_lds[i] = (_Float16)0.f;
  asm volatile("s_wait_dscnt 0x0" ::: "memory");
  __syncthreads();   // all bstage reads done before async x writes reuse it

  // ---------------- async x staging ---------------------------------------
  // chunk cidx: 16 rows x (CH*INQ*4 = 1024 B contiguous per row)
  auto issue = [&](int cidx) {
    const unsigned sel = (unsigned)(cidx & 1);
    const unsigned long long gx =
        (unsigned long long)(size_t)x +
        ((unsigned long long)b0 * TT + (unsigned long long)cidx * CH) * (INQ * 4);
    const unsigned lb = ldsbase + sel * 16384u;
    #pragma unroll
    for (int i = 0; i < 8; ++i) {
      int idx = tid + i * 128;
      int row = idx >> 6;
      int o16 = idx & 63;
      unsigned long long ga = gx + (unsigned long long)row * (TT * INQ * 4)
                                 + (unsigned)(o16 * 16);
      unsigned la = lb + (unsigned)(row * 1024 + o16 * 16);
      asm volatile("global_load_async_to_lds_b128 %0, %1, off"
                   :: "v"(la), "v"(ga) : "memory");
    }
  };
  issue(0);

  const int m     = lane & 15;
  const int kbase = (lane < 16) ? 0 : 8;     // A-fragment K striping base
  float cst[8] = {0.f, 0.f, 0.f, 0.f, 0.f, 0.f, 0.f, 0.f};
  float hreg[8];

  // Persistent a2 = [x | 1 | 0] fragment: zeros + bias column written once;
  // only e[0..7] (the x values, lanes 0..15) are refreshed each step.
  AF a2;
  #pragma unroll
  for (int j = 0; j < 16; ++j) a2.e[j] = (_Float16)0.f;
  if (lane >= 16) a2.e[0] = (_Float16)1.f;   // K(local) 8 = bias column

  const v8f z = {0.f, 0.f, 0.f, 0.f, 0.f, 0.f, 0.f, 0.f};

  for (int c = 0; c < NC; ++c) {
    const float* xb = (const float*)(smem + (c & 1) * 16384);
    if (c + 1 < NC) {
      issue(c + 1);
      asm volatile("s_wait_asynccnt 0x8" ::: "memory");  // chunk c retired
    } else {
      asm volatile("s_wait_asynccnt 0x0" ::: "memory");
    }
    __syncthreads();

    for (int tl = 0; tl < CH; ++tl) {
      // ---- A fragments: a0/a1 = h cols [0,32)/[32,64); a2 = [x | 1 | 0]
      AF a0, a1;
      const _Float16* hp = h_lds + m * 64;
      a0.q[0] = *(const f32x4*)(hp + kbase);
      a0.q[1] = *(const f32x4*)(hp + kbase + 16);
      a1.q[0] = *(const f32x4*)(hp + 32 + kbase);
      a1.q[1] = *(const f32x4*)(hp + 32 + kbase + 16);
      if (lane < 16) {                       // K(local) 0..7 = x_t[m][:]
        const float* xp = xb + (m * CH + tl) * 8;
        f32x4 x0 = *(const f32x4*)xp;
        f32x4 x1 = *(const f32x4*)(xp + 4);
        #pragma unroll
        for (int j = 0; j < 4; ++j) { a2.e[j] = (_Float16)x0[j]; a2.e[4 + j] = (_Float16)x1[j]; }
      }

      // ---- gates[16 x 64] for this wave's gate type: 12 WMMAs,
      // kc-outer order -> 4 independent WMMAs back-to-back per K chunk.
      v8f acc[4];
      #pragma unroll
      for (int nt = 0; nt < 4; ++nt)
        acc[nt] = __builtin_amdgcn_wmma_f32_16x16x32_f16(false, a0.v, false, Bf[0][nt],
                                                         (short)0, z, false, false);
      #pragma unroll
      for (int nt = 0; nt < 4; ++nt)
        acc[nt] = __builtin_amdgcn_wmma_f32_16x16x32_f16(false, a1.v, false, Bf[1][nt],
                                                         (short)0, acc[nt], false, false);
      #pragma unroll
      for (int nt = 0; nt < 4; ++nt)
        acc[nt] = __builtin_amdgcn_wmma_f32_16x16x32_f16(false, a2.v, false, Bf[2][nt],
                                                         (short)0, acc[nt], false, false);

      float* gbase = gate_lds + wid * 1024 + (lane & 15) * 16 + ((lane < 16) ? 0 : 8);
      #pragma unroll
      for (int nt = 0; nt < 4; ++nt) {
        float* gp = gbase + nt * 256;        // [gate][n][m], m contiguous
        #pragma unroll
        for (int r = 0; r < 8; r += 2) {
          f32x2 t2 = { acc[nt][r], acc[nt][r + 1] };
          *(f32x2*)(gp + r) = t2;
        }
      }
      __syncthreads();

      // ---- elementwise: this wave owns h columns [wid*16, wid*16+16)
      {
        const int nloc = wid * 16 + (lane >> 1);
        const int mb   = (lane & 1) * 8;
        const float* gI = gate_lds + 0 * 1024 + nloc * 16 + mb;
        const float* gF = gate_lds + 1 * 1024 + nloc * 16 + mb;
        const float* gG = gate_lds + 2 * 1024 + nloc * 16 + mb;
        const float* gO = gate_lds + 3 * 1024 + nloc * 16 + mb;
        float iv[8], fv[8], gv[8], ov[8];
        *(f32x4*)&iv[0] = *(const f32x4*)gI; *(f32x4*)&iv[4] = *(const f32x4*)(gI + 4);
        *(f32x4*)&fv[0] = *(const f32x4*)gF; *(f32x4*)&fv[4] = *(const f32x4*)(gF + 4);
        *(f32x4*)&gv[0] = *(const f32x4*)gG; *(f32x4*)&gv[4] = *(const f32x4*)(gG + 4);
        *(f32x4*)&ov[0] = *(const f32x4*)gO; *(f32x4*)&ov[4] = *(const f32x4*)(gO + 4);
        _Float16* hw = h_lds + nloc;
        #pragma unroll
        for (int e2 = 0; e2 < 8; ++e2) {
          float ii = sigm(iv[e2]);
          float ff = sigm(fv[e2]);
          float gg = tanh_hw(gv[e2]);
          float oo = sigm(ov[e2]);
          cst[e2] = ff * cst[e2] + ii * gg;
          float hh = oo * tanh_hw(cst[e2]);
          hreg[e2] = hh;
          hw[(mb + e2) * 64] = (_Float16)hh;
        }
      }
      __syncthreads();
    }
  }

  // ---------------- final FC: out[b] = sum_n h[b][n] * W_fc[n] + b_fc ------
  if (tid < 16) outred[tid] = 0.f;
  __syncthreads();
  {
    const int   nloc = wid * 16 + (lane >> 1);
    const int   mb   = (lane & 1) * 8;
    const float wf   = W_fc[nloc];
    #pragma unroll
    for (int e2 = 0; e2 < 8; ++e2)
      atomicAdd(&outred[mb + e2], hreg[e2] * wf);   // ds_add_f32
  }
  __syncthreads();
  if (tid < 16) out[b0 + tid] = outred[tid] + b_fc[0];
}

extern "C" void kernel_launch(void* const* d_in, const int* in_sizes, int n_in,
                              void* d_out, int out_size, void* d_ws, size_t ws_size,
                              hipStream_t stream) {
  (void)in_sizes; (void)n_in; (void)out_size; (void)d_ws; (void)ws_size;
  const float* x    = (const float*)d_in[0];
  const float* W_ih = (const float*)d_in[1];
  const float* W_hh = (const float*)d_in[2];
  const float* b_ih = (const float*)d_in[3];
  const float* b_hh = (const float*)d_in[4];
  const float* W_fc = (const float*)d_in[5];
  const float* b_fc = (const float*)d_in[6];
  float* out = (float*)d_out;

  dim3 grid(BQ / 16), block(128);
  hipLaunchKernelGGL(solar_lstm_wmma, grid, block, 0, stream,
                     x, W_ih, W_hh, b_ih, b_hh, W_fc, b_fc, out);
}
